// LateralLayer_12068858102064
// MI455X (gfx1250) — matrix operand order repaired and checked by reading, not compiled, not run on hardware
//
#include <hip/hip_runtime.h>
#include <hip/hip_bf16.h>

typedef __attribute__((ext_vector_type(16))) _Float16 v16h;
typedef __attribute__((ext_vector_type(8)))  float    v8f;

typedef unsigned int       uint32;
typedef unsigned long long uint64;

#define C_IN   68
#define C_OUT  64
#define HW     64
#define B_SZ   16
#define KTAPS  25               // 5x5 taps per channel
#define SLABW  68               // 64 + 2*2 halo columns (cols 0,1,66,67 always zero)
#define SLABH  6                // two output rows -> 6 halo rows
#define SLABCH (SLABH * SLABW)  // halves per channel in the slab

// ---------------------------------------------------------------------------
// Tap permutation: WMMA K-slot -> tap index (j*5+k), chosen so each lane's
// dword-aligned A slot-pairs are row-adjacent in the LDS slab (see BUILD_A).
// ---------------------------------------------------------------------------
__device__ __constant__ int SLOT2TAP[32] = {
     0,  1,  2,  3,      // s0..3   row0 k0..3
    10, 11, 12, 13,      // s4..7   row2 k0..3
     5,  6,  7,  8,      // s8..11  row1 k0..3
    15, 16, 17, 18,      // s12..15 row3 k0..3
    20, 21, 22, 23,      // s16..19 row4 k0..3
    24, -1, 19, -1,      // s20..23 row4 k4, Z, row3 k4, Z
     4, -1,  9, -1,      // s24..27 row0 k4, Z, row1 k4, Z
    14, -1, -1, -1       // s28..31 row2 k4, Z, Z, Z
};

// ---------------------------------------------------------------------------
// Prep: pack W_lateral (64 x 1700 fp32) into WMMA B fragments, hi/lo f16
// split.  Layout: Bpack[c][pass][nt][lane][e]; B column n = lane%16
// (o = nt*16 + n), K slot = (e&7) + 8*(lane/16) + 16*(e/8), tap = SLOT2TAP[slot].
// ---------------------------------------------------------------------------
__global__ void prep_weights(const float* __restrict__ Wl,
                             _Float16* __restrict__ Bp) {
    int idx = blockIdx.x * blockDim.x + threadIdx.x;
    const int total = C_IN * 2 * 4 * 32 * 16;
    if (idx >= total) return;
    int e    = idx & 15;
    int lane = (idx >> 4) & 31;
    int nt   = (idx >> 9) & 3;
    int p    = (idx >> 11) & 1;
    int c    = idx >> 12;
    int o    = nt * 16 + (lane & 15);
    int slot = (e & 7) + (lane >> 4) * 8 + (e >> 3) * 16;
    int tap  = SLOT2TAP[slot];
    float w  = (tap >= 0) ? Wl[o * (C_IN * KTAPS) + c * KTAPS + tap] : 0.0f;
    _Float16 hi = (_Float16)w;
    Bp[idx] = (p == 0) ? hi : (_Float16)(w - (float)hi);
}

// --------------------------- small LDS helpers -----------------------------
__device__ __forceinline__ uint32 lds_u16z(const _Float16* p) {
    unsigned short s; __builtin_memcpy(&s, p, 2); return (uint32)s;
}
__device__ __forceinline__ uint32 lds_u32(const _Float16* p) {
    uint32 r; __builtin_memcpy(&r, p, 4); return r;
}
__device__ __forceinline__ uint64 lds_u64(const _Float16* p) {
    uint64 r; __builtin_memcpy(&r, p, 8); return r;
}
__device__ __forceinline__ uint32 pack2h(float a, float b) {
    _Float16 ha = (_Float16)a, hb = (_Float16)b;
    unsigned short ua, ub;
    __builtin_memcpy(&ua, &ha, 2); __builtin_memcpy(&ub, &hb, 2);
    return (uint32)ua | ((uint32)ub << 16);
}

// ---------------------------------------------------------------------------
// Main conv kernel: one workgroup per (b, h0..h0+1) row pair, 4 waves.
// Wave wv: output row hr = h0 + (wv>>1), pixels w = (wv&1)*32 .. +31 as two
// 16-pixel A-subtiles; all 64 output channels.  Each B fragment feeds 2
// WMMAs (A0, A1) -> 16 WMMAs per channel per wave for 16 b128 loads.
// ---------------------------------------------------------------------------
__launch_bounds__(128)
__global__ void lateral_wmma(const float* __restrict__ x,
                             const _Float16* __restrict__ Bp,
                             float* __restrict__ out) {
    __shared__ __attribute__((aligned(16))) _Float16 xs[C_IN * SLABCH];

    const int tid = threadIdx.x;
    const int b   = blockIdx.x >> 5;          // / 32
    const int h0  = (blockIdx.x & 31) * 2;

    // ---- Phase 1a: zero the w-halo columns (i = 0,1 and 66,67) ----
    {
        uint32* xw = (uint32*)xs;
        for (int q = tid; q < C_IN * SLABH * 2; q += 128) {
            int c = q / (SLABH * 2);
            int r = q - c * (SLABH * 2);
            int j = r >> 1;
            int i = (r & 1) ? 66 : 0;
            xw[(c * SLABCH + j * SLABW + i) >> 1] = 0u;
        }
    }
    // ---- Phase 1b: interior columns, float4 loads -> packed f16 stores ----
    {
        uint32* xw = (uint32*)xs;
        for (int q = tid; q < C_IN * SLABH * 16; q += 128) {
            int c = q / (SLABH * 16);
            int r = q - c * (SLABH * 16);
            int j = r >> 4;
            int m = r & 15;                    // 4-col chunk: wg = 4m .. 4m+3
            int hg = h0 + j - 2;
            float4 v = make_float4(0.f, 0.f, 0.f, 0.f);
            if (hg >= 0 && hg < HW)
                v = *(const float4*)(x + (((b * C_IN + c) * HW + hg) << 6) + m * 4);
            int off = c * SLABCH + j * SLABW + m * 4 + 2;   // even
            xw[off >> 1]       = pack2h(v.x, v.y);
            xw[(off >> 1) + 1] = pack2h(v.z, v.w);
        }
    }
    __syncthreads();

    const int lane    = tid & 31;
    const int wv      = tid >> 5;
    const int halfsel = lane >> 4;
    const int mrow    = lane & 15;            // A matrix row / D column within subtile
    const int jbase   = wv >> 1;              // which output row of the pair
    const int wbase   = (wv & 1) * 32;        // which w-half
    const int wl      = wbase + mrow;         // subtile 0 pixel column

#define R(r_) ((jbase + (r_)) * SLABW)
    // Per-lane A-build offsets (halves, relative to channel base), subtile 0
    const int offq0 = R(halfsel) + wl;                 // rows 0/1, k0..3
    const int offq1 = R(2 + halfsel) + wl;             // rows 2/3, k0..3
    const int off4  = halfsel ? (R(0) + wl + 4) : (R(4) + wl);
    const int off5  = halfsel ? (R(1) + wl + 4) : (R(4) + wl + 2);
    const int off6  = (halfsel ? R(2) : R(4)) + wl + 4;
    const int off7  = halfsel ? R(0) : (R(3) + wl + 4);   // col0 always 0
    const uint32 msk = halfsel ? 0x0000FFFFu : 0xFFFFFFFFu;
#undef R

    const v16h* __restrict__ Bf = (const v16h*)Bp;

    v8f acc[2][4] = {{v8f{}, v8f{}, v8f{}, v8f{}},
                     {v8f{}, v8f{}, v8f{}, v8f{}}};

#define BUILD_A(A_, cbase_, s_)                                              \
    do {                                                                     \
        const _Float16* cb_ = xs + (cbase_) + (s_) * 16;                     \
        (A_).q[0] = lds_u64(cb_ + offq0);                                    \
        (A_).q[1] = lds_u64(cb_ + offq1);                                    \
        (A_).u[4] = lds_u32(cb_ + off4) & msk;                               \
        (A_).u[5] = lds_u32(cb_ + off5) & msk;                               \
        (A_).u[6] = lds_u16z(cb_ + off6);                                    \
        (A_).u[7] = lds_u16z(cb_ + off7);                                    \
    } while (0)

    union AFrag { v16h v; uint32 u[8]; uint64 q[4]; };

    for (int c = 0; c < C_IN; ++c) {
        AFrag A0, A1;
        BUILD_A(A0, c * SLABCH, 0);
        BUILD_A(A1, c * SLABCH, 1);

        v16h bb[8];
#pragma unroll
        for (int f = 0; f < 8; ++f)
            bb[f] = Bf[(c * 8 + f) * 32 + lane];

#pragma unroll
        for (int p = 0; p < 2; ++p) {
#pragma unroll
            for (int nt = 0; nt < 4; ++nt) {
                acc[0][nt] = __builtin_amdgcn_wmma_f32_16x16x32_f16(
                    false, A0.v, false, bb[p * 4 + nt], (short)0,
                    acc[0][nt], false, false);
                acc[1][nt] = __builtin_amdgcn_wmma_f32_16x16x32_f16(
                    false, A1.v, false, bb[p * 4 + nt], (short)0,
                    acc[1][nt], false, false);
            }
        }
    }

    // ---- Writeback per CDNA5 C/D layout: N = lane%16, M = r + 8*(lane/16) ----
    const int hr = h0 + jbase;
#pragma unroll
    for (int s = 0; s < 2; ++s) {
#pragma unroll
        for (int nt = 0; nt < 4; ++nt) {
            const int o = nt * 16 + mrow;
            float* orow = out + (((b * C_OUT + o) * HW + hr) << 6);
#pragma unroll
            for (int r = 0; r < 8; r += 2) {
                float2 v2 = make_float2(acc[s][nt][r], acc[s][nt][r + 1]);
                *(float2*)&orow[wbase + s * 16 + r + halfsel * 8] = v2;
            }
        }
    }
#undef BUILD_A
}

// ---------------------------------------------------------------------------
extern "C" void kernel_launch(void* const* d_in, const int* in_sizes, int n_in,
                              void* d_out, int out_size, void* d_ws, size_t ws_size,
                              hipStream_t stream) {
    const float* x  = (const float*)d_in[0];   // (16, 68, 64, 64) fp32
    const float* Wl = (const float*)d_in[1];   // (64, 1700) fp32
    float* outp     = (float*)d_out;           // (16, 64, 64, 64) fp32
    _Float16* Bpack = (_Float16*)d_ws;         // 68*2*4*32*16 halves = 557 KB

    const int prep_total = C_IN * 2 * 4 * 32 * 16;
    prep_weights<<<(prep_total + 255) / 256, 256, 0, stream>>>(Wl, Bpack);

    lateral_wmma<<<B_SZ * (HW / 2), 128, 0, stream>>>(x, Bpack, outp);
}
